// ONLSTMStack_8546984919641
// MI455X (gfx1250) — matrix operand order repaired and checked by reading, not compiled, not run on hardware
//
#include <hip/hip_runtime.h>
#include <cstdint>
#include <cstddef>

typedef __attribute__((ext_vector_type(16))) _Float16 v16h;
typedef __attribute__((ext_vector_type(8)))  _Float16 v8h;
typedef __attribute__((ext_vector_type(8)))  float    v8f;
typedef __attribute__((ext_vector_type(4)))  int      v4i;

#define BATCH  64
#define SEQLEN 256
#define NCHUNK 16
#define MROWS  (BATCH*SEQLEN)   /* 16384 */
#define KCHUNK 64

#define TRY_ASYNC_LDS 1
#if TRY_ASYNC_LDS && __has_builtin(__builtin_amdgcn_global_load_async_to_lds_b128) && __has_builtin(__builtin_amdgcn_s_wait_asynccnt)
#define HAVE_ASYNC_LDS 1
#else
#define HAVE_ASYNC_LDS 0
#endif

__device__ __forceinline__ float sigf(float x){ return 1.0f/(1.0f+__expf(-x)); }

// ---------------- elementwise helpers ----------------
__global__ void k_f32_to_f16(const float* __restrict__ in, _Float16* __restrict__ out, int n){
  int i = blockIdx.x*blockDim.x + threadIdx.x;
  if(i<n) out[i] = (_Float16)in[i];
}
__global__ void k_zero_f16(_Float16* __restrict__ p, int n){
  int i = blockIdx.x*blockDim.x + threadIdx.x;
  if(i<n) p[i] = (_Float16)0.0f;
}
__global__ void k_zero_f32(float* __restrict__ p, int n){
  int i = blockIdx.x*blockDim.x + threadIdx.x;
  if(i<n) p[i] = 0.0f;
}
// (B,L,D) f32 -> (L,B,D) f16
__global__ void k_tr_convert(const float* __restrict__ in, _Float16* __restrict__ out, int D, int total){
  int i = blockIdx.x*blockDim.x + threadIdx.x;
  if(i>=total) return;
  int d = i % D;
  int r = i / D;          // r = l*BATCH + b
  int b = r % BATCH;
  int l = r / BATCH;
  out[i] = (_Float16)in[((size_t)b*SEQLEN + l)*D + d];
}

// ---------------- WMMA fragment loaders ----------------
// A: M x K f16 row-major, 16x32 tile: lane row = lane&15, K halves split at lane 16.
__device__ __forceinline__ v16h load_afrag(const _Float16* __restrict__ A, int K, int mrow, int k0, int lane){
  const _Float16* ap = A + (size_t)mrow*K + k0 + ((lane>>4)<<4);
  v8h lo = *reinterpret_cast<const v8h*>(ap);
  v8h hi = *reinterpret_cast<const v8h*>(ap + 8);
  v16h a;
#pragma unroll
  for(int i=0;i<8;i++){ a[i]=lo[i]; a[8+i]=hi[i]; }
  return a;
}
// B fragment from LDS-staged W tile: sW is [64 rows][KCHUNK] row-major f16.
__device__ __forceinline__ v16h load_bfrag_lds(const _Float16* sW, int nrow, int kk, int lane){
  return *reinterpret_cast<const v16h*>(sW + nrow*KCHUNK + kk + ((lane>>4)<<4));
}

// Stage W rows [n0, n0+64) x K cols [k0, k0+KCHUNK) into LDS (8 KB), 128 threads.
__device__ __forceinline__ void stage_w(const _Float16* __restrict__ W, int K, int n0, int k0,
                                        _Float16* sW, int tid)
{
#if HAVE_ASYNC_LDS
  typedef __attribute__((address_space(1))) v4i* gp_t;   // global (AS1) int4*
  typedef __attribute__((address_space(3))) v4i* lp_t;   // LDS (AS3) int4*
#pragma unroll
  for(int p=0;p<4;p++){
    int fidx = p*1024 + tid*8;            // f16 element index within 64x64 tile
    int n = fidx >> 6;
    int k = fidx & 63;
    const _Float16* src = W + (size_t)(n0+n)*K + k0 + k;
    _Float16* dst = sW + n*KCHUNK + k;
    __builtin_amdgcn_global_load_async_to_lds_b128(
        (gp_t)(uintptr_t)src,
        (lp_t)(unsigned)(uintptr_t)dst,
        0, 0);
  }
  __builtin_amdgcn_s_wait_asynccnt(0);
#else
#pragma unroll
  for(int p=0;p<4;p++){
    int fidx = p*1024 + tid*8;
    int n = fidx >> 6;
    int k = fidx & 63;
    *reinterpret_cast<v8h*>(sW + n*KCHUNK + k) =
        *reinterpret_cast<const v8h*>(W + (size_t)(n0+n)*K + k0 + k);
  }
#endif
}

// Core accumulate: acc[4] over full K with LDS-staged B tiles.
__device__ __forceinline__ void gemm_core(const _Float16* __restrict__ A,
                                          const _Float16* __restrict__ W,
                                          _Float16* sW, int K, int mrow, int n0,
                                          int tid, int lane, v8f acc[4])
{
  for(int k0=0;k0<K;k0+=KCHUNK){
    stage_w(W, K, n0, k0, sW, tid);
    __syncthreads();
    if (k0 + KCHUNK < K)
      __builtin_prefetch(A + (size_t)mrow*K + k0 + KCHUNK, 0, 1);
#pragma unroll
    for(int kk=0;kk<KCHUNK;kk+=32){
      v16h a  = load_afrag(A, K, mrow, k0 + kk, lane);
      v16h b0 = load_bfrag_lds(sW, (lane&15) +  0, kk, lane);
      v16h b1 = load_bfrag_lds(sW, (lane&15) + 16, kk, lane);
      v16h b2 = load_bfrag_lds(sW, (lane&15) + 32, kk, lane);
      v16h b3 = load_bfrag_lds(sW, (lane&15) + 48, kk, lane);
      acc[0] = __builtin_amdgcn_wmma_f32_16x16x32_f16(false, a, false, b0, (short)0, acc[0], false, false);
      acc[1] = __builtin_amdgcn_wmma_f32_16x16x32_f16(false, a, false, b1, (short)0, acc[1], false, false);
      acc[2] = __builtin_amdgcn_wmma_f32_16x16x32_f16(false, a, false, b2, (short)0, acc[2], false, false);
      acc[3] = __builtin_amdgcn_wmma_f32_16x16x32_f16(false, a, false, b3, (short)0, acc[3], false, false);
    }
    __syncthreads();
  }
}

// ---------------- input projection GEMM ----------------
// C[m][n] (f16) = sum_k A[m][k]*W[n][k] + bias[n]; M mult 64, N mult 64, K mult 64.
__global__ void k_gemm_in(const _Float16* __restrict__ A, const _Float16* __restrict__ W,
                          const float* __restrict__ bias, _Float16* __restrict__ C,
                          int N, int K)
{
  __shared__ _Float16 sW[64*KCHUNK];
  const int tid  = threadIdx.x;
  const int lane = tid & 31;
  const int wave = tid >> 5;
  const int m0 = blockIdx.y*64 + wave*16;
  const int n0 = blockIdx.x*64;
  const int mrow = m0 + (lane & 15);
  v8f acc[4] = {};
  gemm_core(A, W, sW, K, mrow, n0, tid, lane, acc);
  const int mo = (lane>>4)<<3;
#pragma unroll
  for(int t=0;t<4;t++){
    int n = n0 + t*16 + (lane&15);
    float bv = bias[n];
#pragma unroll
    for(int r=0;r<8;r++){
      C[(size_t)(m0+mo+r)*N + n] = (_Float16)(acc[t][r] + bv);
    }
  }
}

// ---------------- recurrent step GEMM (both LSTM streams) ----------------
// G[b][n] (f32) = tin[b][n] + bias[n] + sum_k H[b][k]*W[n][k];  M = 64 fixed.
__global__ void k_gemm_rec(const _Float16* __restrict__ H0, const _Float16* __restrict__ W0,
                           const float* __restrict__ b0, const _Float16* __restrict__ T0, float* __restrict__ G0,
                           const _Float16* __restrict__ H1, const _Float16* __restrict__ W1,
                           const float* __restrict__ b1, const _Float16* __restrict__ T1, float* __restrict__ G1,
                           int N, int K)
{
  __shared__ _Float16 sW[64*KCHUNK];
  const _Float16* H = blockIdx.y ? H1 : H0;
  const _Float16* W = blockIdx.y ? W1 : W0;
  const float*  bias = blockIdx.y ? b1 : b0;
  const _Float16* T = blockIdx.y ? T1 : T0;
  float* G = blockIdx.y ? G1 : G0;

  const int tid  = threadIdx.x;
  const int lane = tid & 31;
  const int wave = tid >> 5;
  const int m0 = wave*16;            // 4 waves cover the 64 batch rows
  const int n0 = blockIdx.x*64;
  const int mrow = m0 + (lane & 15);
  v8f acc[4] = {};
  gemm_core(H, W, sW, K, mrow, n0, tid, lane, acc);
  const int mo = (lane>>4)<<3;
#pragma unroll
  for(int t=0;t<4;t++){
    int n = n0 + t*16 + (lane&15);
    float bv = bias[n];
#pragma unroll
    for(int r=0;r<8;r++){
      int m = m0 + mo + r;
      G[(size_t)m*N + n] = acc[t][r] + (float)T[(size_t)m*N + n] + bv;
    }
  }
}

// ---------------- ON-LSTM cell (both streams) ----------------
__global__ void k_cell(const float* __restrict__ Gly, const float* __restrict__ Gtn,
                       float* __restrict__ cly, float* __restrict__ ctn,
                       _Float16* __restrict__ outly, _Float16* __restrict__ outtn,
                       const float* __restrict__ emb, const int* __restrict__ sent,
                       int t, int hidden, int chunk)
{
  __shared__ float s_ifl[NCHUNK], s_ff[NCHUNK], s_cin[NCHUNK], s_cfg[NCHUNK];
  const int b = blockIdx.x;
  if(threadIdx.x == 0){
    int idx = sent[b*SEQLEN + t];
    for(int k=0;k<NCHUNK;k++) s_ifl[k] = emb[idx*NCHUNK + k];
    if(t > 0){
      int ip = sent[b*SEQLEN + t - 1];
      for(int k=0;k<NCHUNK;k++) s_ff[k] = emb[ip*NCHUNK + k];
    } else {
      for(int k=0;k<NCHUNK;k++) s_ff[k] = 0.0f;
    }
    auto cumsm = [](const float* v, float* o){   // cumsum(softmax(v)) over 8
      float mx = v[0];
      for(int j=1;j<8;j++) mx = fmaxf(mx, v[j]);
      float e[8]; float s = 0.0f;
      for(int j=0;j<8;j++){ e[j] = __expf(v[j]-mx); s += e[j]; }
      float run = 0.0f;
      for(int j=0;j<8;j++){ run += e[j]/s; o[j] = run; }
    };
    float tmp[8], tout[8];
    cumsm(s_ifl, tout);                              // cin[0:8] = cumsoftmax(hi)
    for(int j=0;j<8;j++) s_cin[j] = tout[j];
    for(int j=0;j<8;j++) tmp[j] = s_ifl[8 + 7 - j];  // cin[8:16] = flip(cumsoftmax(flip(si)))
    cumsm(tmp, tout);
    for(int j=0;j<8;j++) s_cin[8+j] = tout[7-j];
    for(int j=0;j<8;j++) tmp[j] = s_ff[7-j];         // cfg[0:8] = flip(cumsoftmax(flip(hf)))
    cumsm(tmp, tout);
    for(int j=0;j<8;j++) s_cfg[j] = tout[7-j];
    cumsm(s_ff + 8, tout);                           // cfg[8:16] = cumsoftmax(sf)
    for(int j=0;j<8;j++) s_cfg[8+j] = tout[j];
  }
  __syncthreads();
  const int N4 = 4*hidden;
  for(int j = threadIdx.x; j < hidden; j += blockDim.x){
    int k = j / chunk;
    { // lyric stream: gated cin/cfg
      const float* G = Gly + (size_t)b*N4;
      float o  = sigf(G[j]);
      float c  = tanhf(G[hidden + j]);
      float ii = sigf(G[2*hidden + j]);
      float f  = sigf(G[3*hidden + j]);
      float cink = s_cin[k], cfgk = s_cfg[k];
      float ov = cfgk*cink;
      float f2 = f*ov + (cfgk - ov);
      float i2 = ii*ov + (cink - ov);
      float cy = f2*cly[(size_t)b*hidden + j] + i2*c;
      float hy = o*tanhf(cy);
      cly[(size_t)b*hidden + j] = cy;
      outly[(size_t)b*hidden + j] = (_Float16)hy;
    }
    { // tune stream: raw ifl/ff
      const float* G = Gtn + (size_t)b*N4;
      float o  = sigf(G[j]);
      float c  = tanhf(G[hidden + j]);
      float ii = sigf(G[2*hidden + j]);
      float f  = sigf(G[3*hidden + j]);
      float cink = s_ifl[k], cfgk = s_ff[k];
      float ov = cfgk*cink;
      float f2 = f*ov + (cfgk - ov);
      float i2 = ii*ov + (cink - ov);
      float cy = f2*ctn[(size_t)b*hidden + j] + i2*c;
      float hy = o*tanhf(cy);
      ctn[(size_t)b*hidden + j] = cy;
      outtn[(size_t)b*hidden + j] = (_Float16)hy;
    }
  }
}

// ---------------- final MLP + softmax + transpose ----------------
__global__ void k_mlp(const _Float16* __restrict__ X, const _Float16* __restrict__ TX,
                      const float* __restrict__ W, const float* __restrict__ Bv,
                      float* __restrict__ out, int h)
{
  int m = blockIdx.x*blockDim.x + threadIdx.x;   // m = l*BATCH + b
  if(m >= MROWS) return;
  int l = m / BATCH, b = m % BATCH;
  const _Float16* x  = X  + (size_t)m*h;
  const _Float16* tx = TX + (size_t)m*h;
  float logit[4];
#pragma unroll
  for(int o=0;o<4;o++){
    const float* w0 = W + (size_t)o*2*h;
    float acc = Bv[o];
    for(int f=0; f<h; ++f) acc += (float)x[f]*w0[f];
    for(int f=0; f<h; ++f) acc += (float)tx[f]*w0[h+f];
    logit[o] = acc;
  }
  float mx = fmaxf(fmaxf(logit[0],logit[1]), fmaxf(logit[2],logit[3]));
  float e[4]; float s = 0.0f;
#pragma unroll
  for(int o=0;o<4;o++){ e[o] = __expf(logit[o]-mx); s += e[o]; }
#pragma unroll
  for(int o=0;o<4;o++) out[((size_t)b*SEQLEN + l)*4 + o] = e[o]/s;
}

// ---------------- host orchestration ----------------
extern "C" void kernel_launch(void* const* d_in, const int* in_sizes, int n_in,
                              void* d_out, int out_size, void* d_ws, size_t ws_size,
                              hipStream_t stream)
{
  (void)in_sizes; (void)n_in; (void)out_size; (void)ws_size;
  const float* lyric = (const float*)d_in[0];
  const float* tune  = (const float*)d_in[1];
  const float* emb   = (const float*)d_in[2];
  const float* ih_w[2]  = {(const float*)d_in[3],  (const float*)d_in[4]};
  const float* ih_b[2]  = {(const float*)d_in[5],  (const float*)d_in[6]};
  const float* hh_w[2]  = {(const float*)d_in[7],  (const float*)d_in[8]};
  const float* hh_b[2]  = {(const float*)d_in[9],  (const float*)d_in[10]};
  const float* tih_w[2] = {(const float*)d_in[11], (const float*)d_in[12]};
  const float* tih_b[2] = {(const float*)d_in[13], (const float*)d_in[14]};
  const float* thh_w[2] = {(const float*)d_in[15], (const float*)d_in[16]};
  const float* thh_b[2] = {(const float*)d_in[17], (const float*)d_in[18]};
  const float* mlp_w = (const float*)d_in[19];
  const float* mlp_b = (const float*)d_in[20];
  const int*   sent  = (const int*)d_in[21];

  static const int LS[3] = {512, 1024, 512};

  uint8_t* base = (uint8_t*)d_ws;
  size_t off = 0;
  auto alloc = [&](size_t bytes)->void* {
    off = (off + 255) & ~(size_t)255;
    void* p = base + off;
    off += bytes;
    return p;
  };
  auto cvt = [&](const float* src, _Float16* dst, int n){
    k_f32_to_f16<<<(n+255)/256, 256, 0, stream>>>(src, dst, n);
  };

  // f16 weight copies
  _Float16 *w_ih[2], *w_hh[2], *w_tih[2], *w_thh[2];
  for(int l=0;l<2;l++){
    int din = LS[l], h = LS[l+1], n4 = 4*h;
    w_ih[l]  = (_Float16*)alloc((size_t)n4*din*2);
    w_hh[l]  = (_Float16*)alloc((size_t)n4*h*2);
    w_tih[l] = (_Float16*)alloc((size_t)n4*din*2);
    w_thh[l] = (_Float16*)alloc((size_t)n4*h*2);
    cvt(ih_w[l],  w_ih[l],  n4*din);
    cvt(hh_w[l],  w_hh[l],  n4*h);
    cvt(tih_w[l], w_tih[l], n4*din);
    cvt(thh_w[l], w_thh[l], n4*h);
  }

  // transposed f16 inputs (L,B,512)
  _Float16* in_ly = (_Float16*)alloc((size_t)MROWS*LS[0]*2);
  _Float16* in_tn = (_Float16*)alloc((size_t)MROWS*LS[0]*2);
  {
    int tot = MROWS*LS[0];
    k_tr_convert<<<(tot+255)/256, 256, 0, stream>>>(lyric, in_ly, LS[0], tot);
    k_tr_convert<<<(tot+255)/256, 256, 0, stream>>>(tune,  in_tn, LS[0], tot);
  }

  // sequence buffers with L+1 slots (slot 0 = h_{-1} = 0)
  _Float16 *seq_ly[2], *seq_tn[2];
  for(int l=0;l<2;l++){
    int h = LS[l+1];
    seq_ly[l] = (_Float16*)alloc((size_t)(SEQLEN+1)*BATCH*h*2);
    seq_tn[l] = (_Float16*)alloc((size_t)(SEQLEN+1)*BATCH*h*2);
  }
  _Float16* tin_ly = (_Float16*)alloc((size_t)MROWS*4096*2);
  _Float16* tin_tn = (_Float16*)alloc((size_t)MROWS*4096*2);
  float* G_ly = (float*)alloc((size_t)BATCH*4096*4);
  float* G_tn = (float*)alloc((size_t)BATCH*4096*4);
  float* cx   = (float*)alloc((size_t)BATCH*1024*4);
  float* tcx  = (float*)alloc((size_t)BATCH*1024*4);

  const _Float16* curly = in_ly;
  const _Float16* curtn = in_tn;

  for(int l=0;l<2;l++){
    int din = LS[l], h = LS[l+1], n4 = 4*h, chunk = h/NCHUNK;

    // input projections over all timesteps (big WMMA GEMMs)
    dim3 gin(n4/64, MROWS/64);
    k_gemm_in<<<gin, 128, 0, stream>>>(curly, w_ih[l],  ih_b[l],  tin_ly, n4, din);
    k_gemm_in<<<gin, 128, 0, stream>>>(curtn, w_tih[l], tih_b[l], tin_tn, n4, din);

    // zero initial hidden/cell states
    int hs = BATCH*h;
    k_zero_f16<<<(hs+255)/256, 256, 0, stream>>>(seq_ly[l], hs);
    k_zero_f16<<<(hs+255)/256, 256, 0, stream>>>(seq_tn[l], hs);
    k_zero_f32<<<(hs+255)/256, 256, 0, stream>>>(cx,  hs);
    k_zero_f32<<<(hs+255)/256, 256, 0, stream>>>(tcx, hs);

    // sequential recurrence
    for(int t=0;t<SEQLEN;t++){
      dim3 gr(n4/64, 2);
      k_gemm_rec<<<gr, 128, 0, stream>>>(
          seq_ly[l] + (size_t)t*BATCH*h, w_hh[l],  hh_b[l],  tin_ly + (size_t)t*BATCH*n4, G_ly,
          seq_tn[l] + (size_t)t*BATCH*h, w_thh[l], thh_b[l], tin_tn + (size_t)t*BATCH*n4, G_tn,
          n4, h);
      k_cell<<<BATCH, 256, 0, stream>>>(
          G_ly, G_tn, cx, tcx,
          seq_ly[l] + (size_t)(t+1)*BATCH*h,
          seq_tn[l] + (size_t)(t+1)*BATCH*h,
          emb, sent, t, h, chunk);
    }
    curly = seq_ly[l] + BATCH*h;   // slots 1..L are this layer's outputs, contiguous
    curtn = seq_tn[l] + BATCH*h;
  }

  k_mlp<<<(MROWS+255)/256, 256, 0, stream>>>(curly, curtn, mlp_w, mlp_b, (float*)d_out, LS[2]);
}
